// ProteinGCN_65841848648263
// MI455X (gfx1250) — matrix operand order repaired and checked by reading, not compiled, not run on hardware
//
#include <hip/hip_runtime.h>

// ---------------------------------------------------------------------------
// ProteinGCN forward on MI455X (gfx1250, wave32).
// GEMMs via v_wmma_f32_16x16x32_bf16 (fp32 accum): 128x64 block tile,
// 4 WMMA per wave per 32-K chunk, double-buffered fragment-ordered LDS
// (one s_barrier per chunk). Scatter via global f32 atomics.
// GCN bias skipped: BN(h+b) == BN(h) exactly (training-mode BN).
// ---------------------------------------------------------------------------

typedef __attribute__((ext_vector_type(16))) __bf16 bf16x16;
typedef __attribute__((ext_vector_type(8)))  float  f32x8;

#define TILE_M 128
#define TILE_N 64

union FragU { uint4 q[2]; bf16x16 v; };

__device__ __forceinline__ unsigned packbf(float a, float b) {
    union { __bf16 h[2]; unsigned u; } p;
    p.h[0] = (__bf16)a;      // low 16 bits  = even element
    p.h[1] = (__bf16)b;      // high 16 bits = odd element
    return p.u;
}

// ------------------------- WMMA GEMM: C = A @ W (+bias) (+relu) -------------
// A:[M,K] fp32 row-major, W:[K,Nout] fp32 row-major. K%32==0, Nout%64==0.
__global__ __launch_bounds__(256)
void gemm_wmma_bf16(const float* __restrict__ A, const float* __restrict__ W,
                    const float* __restrict__ bias, float* __restrict__ C,
                    int M, int K, int Nout, int fuseRelu)
{
    // Fragment-ordered LDS, double buffered:
    // sAf[buf][echunk][mtile*32+lane], sBf[buf][echunk][ctile*32+lane]
    __shared__ uint4 sAf[2][2][256];   // 8 row-tiles x 32 lanes  (16 KB)
    __shared__ uint4 sBf[2][2][128];   // 4 col-tiles x 32 lanes  ( 8 KB)

    const int tid  = threadIdx.x;
    const int lane = tid & 31;
    const int wid  = tid >> 5;           // 8 waves = 8 row tiles of 16
    const int half = lane >> 4;
    const int lm   = lane & 15;

    const int rowBase = blockIdx.x * TILE_M;
    const int colBase = blockIdx.y * TILE_N;

    f32x8 acc[4] = {{}, {}, {}, {}};

    // --- A staging map: thread -> (tile row, 16-wide k half) ----------------
    const int la_r   = tid >> 1;               // 0..127
    const int a_eh   = tid & 1;                // k in [16*a_eh, 16*a_eh+16)
    const int a_slot = (la_r >> 4) * 32 + (la_r & 15);   // +16 for lane-half 1
    const int a_row  = rowBase + la_r;
    const bool a_ok  = (a_row < M);
    const float* aptr = A + (size_t)a_row * K + a_eh * 16;

    // --- B staging map: thread -> (col, k-group of 8) -----------------------
    const int b_c    = tid & 63;               // coalesced across wave
    const int b_kg   = (tid >> 6) * 8;         // 0,8,16,24
    const int b_eh   = (b_kg >> 4);
    const int b_slot = (b_c >> 4) * 32 + (b_c & 15) + 16 * ((b_kg >> 3) & 1);
    const float* bptr = W + (size_t)b_kg * Nout + colBase + b_c;

    const int afidx = wid * 32 + lane;

    auto stage = [&](int kk, int buf) {
        // A: 4x float4 global load (64B of one row) -> 2x ds_store_b128
        float4 f0 = {0,0,0,0}, f1 = {0,0,0,0}, f2 = {0,0,0,0}, f3 = {0,0,0,0};
        if (a_ok) {
            const float4* p = (const float4*)(aptr + kk);
            f0 = p[0]; f1 = p[1]; f2 = p[2]; f3 = p[3];
        }
        uint4 lo, hi;
        lo.x = packbf(f0.x, f0.y); lo.y = packbf(f0.z, f0.w);
        lo.z = packbf(f1.x, f1.y); lo.w = packbf(f1.z, f1.w);
        hi.x = packbf(f2.x, f2.y); hi.y = packbf(f2.z, f2.w);
        hi.z = packbf(f3.x, f3.y); hi.w = packbf(f3.z, f3.w);
        sAf[buf][a_eh][a_slot]      = lo;   // j=0..7  -> lane-half 0
        sAf[buf][a_eh][a_slot + 16] = hi;   // j=8..15 -> lane-half 1
        // B: 8 strided b32 loads (lane-coalesced per k-row) -> 1x ds_store_b128
        const float* p = bptr + (size_t)kk * Nout;
        float v0 = p[0];
        float v1 = p[(size_t)1 * Nout];
        float v2 = p[(size_t)2 * Nout];
        float v3 = p[(size_t)3 * Nout];
        float v4 = p[(size_t)4 * Nout];
        float v5 = p[(size_t)5 * Nout];
        float v6 = p[(size_t)6 * Nout];
        float v7 = p[(size_t)7 * Nout];
        uint4 pk;
        pk.x = packbf(v0, v1); pk.y = packbf(v2, v3);
        pk.z = packbf(v4, v5); pk.w = packbf(v6, v7);
        sBf[buf][b_eh][b_slot] = pk;
    };

    const int nk = K >> 5;
    stage(0, 0);
    __syncthreads();

    for (int i = 0; i < nk; ++i) {
        const int buf = i & 1;
        if (i + 1 < nk) stage((i + 1) << 5, buf ^ 1);   // prefetch next chunk

        FragU fa;
        fa.q[0] = sAf[buf][0][afidx];
        fa.q[1] = sAf[buf][1][afidx];
        #pragma unroll
        for (int t = 0; t < 4; ++t) {
            FragU fb;
            fb.q[0] = sBf[buf][0][t * 32 + lane];
            fb.q[1] = sBf[buf][1][t * 32 + lane];
            acc[t] = __builtin_amdgcn_wmma_f32_16x16x32_bf16(
                         false, fa.v, false, fb.v, (short)0, acc[t], false, false);
        }
        __syncthreads();
    }

    // epilogue: C/D layout -> row = r + 8*half, col = lane&15
    #pragma unroll
    for (int r = 0; r < 8; ++r) {
        int row = rowBase + wid * 16 + r + 8 * half;
        if (row < M) {
            #pragma unroll
            for (int t = 0; t < 4; ++t) {
                int c = colBase + t * 16 + lm;
                float v = acc[t][r] + (bias ? bias[c] : 0.0f);
                if (fuseRelu) v = fmaxf(v, 0.0f);
                C[(size_t)row * Nout + c] = v;
            }
        }
    }
}

// ------------------------- degree / norm ------------------------------------
__global__ void deg_init(float* deg, int N) {
    int i = blockIdx.x * 256 + threadIdx.x;
    if (i < N) deg[i] = 1.0f;                       // self-loop
}
__global__ void deg_accum(const int* __restrict__ dst, float* deg, int E) {
    int e = blockIdx.x * 256 + threadIdx.x;
    if (e < E) atomicAdd(&deg[dst[e]], 1.0f);
}
__global__ void dinv_k(const float* __restrict__ deg, float* dinv, int N) {
    int i = blockIdx.x * 256 + threadIdx.x;
    if (i < N) dinv[i] = rsqrtf(deg[i]);            // deg >= 1 always
}

// ------------------------- scatter (self loop init + edges) -----------------
__global__ void scatter_self(const float* __restrict__ in, const float* __restrict__ dinv,
                             float* __restrict__ out, int N, int dout)
{
    size_t gid = (size_t)blockIdx.x * 256 + threadIdx.x;
    int c4n = dout >> 2;
    size_t n4 = (size_t)N * c4n;
    if (gid >= n4) return;
    int r = (int)(gid / c4n);
    float w = dinv[r] * dinv[r];
    float4 v = ((const float4*)in)[gid];
    v.x *= w; v.y *= w; v.z *= w; v.w *= w;
    ((float4*)out)[gid] = v;
}

__global__ void edge_scatter(const float* __restrict__ msg, const int* __restrict__ src,
                             const int* __restrict__ dst, const float* __restrict__ dinv,
                             float* __restrict__ out, int E, int dout)
{
    size_t gid = (size_t)blockIdx.x * 256 + threadIdx.x;
    int c4n = dout >> 2;
    size_t n = (size_t)E * c4n;
    if (gid >= n) return;
    int e = (int)(gid / c4n);
    int q = (int)(gid % c4n);
    int s = src[e], d = dst[e];
    float w = dinv[s] * dinv[d];
    float4 v = ((const float4*)msg)[(size_t)s * c4n + q];
    float* o = out + (size_t)d * dout + q * 4;
    atomicAdd(o + 0, v.x * w);
    atomicAdd(o + 1, v.y * w);
    atomicAdd(o + 2, v.z * w);
    atomicAdd(o + 3, v.w * w);
}

// ------------------------- BatchNorm (training mode) ------------------------
__global__ void zero_f(float* p, int n) {
    int i = blockIdx.x * 256 + threadIdx.x;
    if (i < n) p[i] = 0.0f;
}

__global__ __launch_bounds__(256)
void bn_stats(const float* __restrict__ h, float* __restrict__ stat, int N, int dout)
{
    __shared__ float s_sum[256];
    __shared__ float s_sq[256];
    int tid = threadIdx.x;
    s_sum[tid] = 0.0f; s_sq[tid] = 0.0f;
    __syncthreads();
    int r = blockIdx.x * 256 + tid;
    if (r < N) {
        const float* row = h + (size_t)r * dout;
        for (int j = 0; j < dout; ++j) {
            int c = (tid + j) & (dout - 1);          // stagger -> few conflicts
            float v = row[c];
            atomicAdd(&s_sum[c], v);
            atomicAdd(&s_sq[c], v * v);
        }
    }
    __syncthreads();
    if (tid < dout) {
        atomicAdd(&stat[tid], s_sum[tid]);
        atomicAdd(&stat[256 + tid], s_sq[tid]);
    }
}

__global__ void bn_finalize(const float* __restrict__ stat, const float* __restrict__ gamma,
                            const float* __restrict__ beta, float* __restrict__ bnsc,
                            int N, int dout)
{
    int c = threadIdx.x;
    if (c < dout) {
        float invN = 1.0f / (float)N;
        float m = stat[c] * invN;
        float var = fmaxf(stat[256 + c] * invN - m * m, 0.0f);   // biased var
        float rstd = rsqrtf(var + 1e-5f);
        float sc = gamma[c] * rstd;
        bnsc[c] = sc;
        bnsc[256 + c] = beta[c] - m * sc;
    }
}

__global__ void bn_apply_relu(float* __restrict__ h, const float* __restrict__ bnsc,
                              int N, int dout)
{
    size_t gid = (size_t)blockIdx.x * 256 + threadIdx.x;
    int c4n = dout >> 2;
    size_t n4 = (size_t)N * c4n;
    if (gid >= n4) return;
    int c4 = (int)(gid % c4n);
    float4 sc = ((const float4*)bnsc)[c4];
    float4 sh = ((const float4*)(bnsc + 256))[c4];
    float4 v = ((float4*)h)[gid];
    v.x = fmaxf(v.x * sc.x + sh.x, 0.0f);
    v.y = fmaxf(v.y * sc.y + sh.y, 0.0f);
    v.z = fmaxf(v.z * sc.z + sh.z, 0.0f);
    v.w = fmaxf(v.w * sc.w + sh.w, 0.0f);
    ((float4*)h)[gid] = v;
}

// ------------------------- global mean pool ---------------------------------
__global__ void pool_acc(const float* __restrict__ h, const int* __restrict__ batch,
                         float* __restrict__ psum, float* __restrict__ pcnt,
                         int N, int dout)
{
    int r = blockIdx.x * 256 + threadIdx.x;
    if (r >= N) return;
    int b = batch[r];
    atomicAdd(&pcnt[b], 1.0f);
    const float* row = h + (size_t)r * dout;
    for (int j = 0; j < dout; ++j) {
        int c = (threadIdx.x + j) & (dout - 1);
        atomicAdd(&psum[(size_t)b * dout + c], row[c]);
    }
}

__global__ void pool_div(const float* __restrict__ psum, const float* __restrict__ pcnt,
                         float* __restrict__ g, int dout)
{
    int gid = blockIdx.x * 256 + threadIdx.x;
    if (gid >= 64 * dout) return;
    int b = gid / dout;
    g[gid] = psum[gid] / fmaxf(pcnt[b], 1.0f);
}

// ---------------------------------------------------------------------------
static inline unsigned nblk(size_t n, int t) { return (unsigned)((n + t - 1) / t); }

extern "C" void kernel_launch(void* const* d_in, const int* in_sizes, int n_in,
                              void* d_out, int out_size, void* d_ws, size_t ws_size,
                              hipStream_t stream)
{
    const int N = 50000, E = 1600000;

    const float* x      = (const float*)d_in[0];
    const int*   eidx   = (const int*)d_in[1];
    const int*   batch  = (const int*)d_in[2];
    const float* Wg[3]    = {(const float*)d_in[3],  (const float*)d_in[7],  (const float*)d_in[11]};
    const float* gamma[3] = {(const float*)d_in[5],  (const float*)d_in[9],  (const float*)d_in[13]};
    const float* beta[3]  = {(const float*)d_in[6],  (const float*)d_in[10], (const float*)d_in[14]};
    const float* Wf0 = (const float*)d_in[15];
    const float* bf0 = (const float*)d_in[16];
    const float* Wf1 = (const float*)d_in[17];
    const float* bf1 = (const float*)d_in[18];
    float* out = (float*)d_out;

    const int* srcp = eidx;         // edge_index[0]
    const int* dstp = eidx + E;     // edge_index[1]

    // workspace carve-up (floats)
    float* ws = (float*)d_ws;
    size_t off = 0;
    float* deg  = ws + off; off += N;
    float* dnv  = ws + off; off += N;
    float* bufA = ws + off; off += (size_t)N * 256;   // pre-aggregation h@W
    float* bufB = ws + off; off += (size_t)N * 256;   // aggregated / activations
    float* stat = ws + off; off += 512;               // sum[256] | sumsq[256]
    float* bnsc = ws + off; off += 512;               // scale[256] | shift[256]
    float* psum = ws + off; off += 64 * 256;
    float* pcnt = ws + off; off += 64;
    float* gbuf = ws + off; off += 64 * 256;
    float* g1   = ws + off; off += 64 * 1024;

    // --- symmetric GCN normalization ---
    deg_init <<<nblk(N, 256), 256, 0, stream>>>(deg, N);
    deg_accum<<<nblk(E, 256), 256, 0, stream>>>(dstp, deg, E);
    dinv_k   <<<nblk(N, 256), 256, 0, stream>>>(deg, dnv, N);

    // --- 3 GCN layers ---
    const int dims[4] = {1280, 128, 256, 256};
    const float* hin = x;
    for (int l = 0; l < 3; ++l) {
        int K = dims[l], D = dims[l + 1];
        dim3 gg(nblk(N, TILE_M), (unsigned)(D / TILE_N));
        gemm_wmma_bf16<<<gg, 256, 0, stream>>>(hin, Wg[l], nullptr, bufA, N, K, D, 0);

        size_t n4 = (size_t)N * (D >> 2);
        scatter_self<<<nblk(n4, 256), 256, 0, stream>>>(bufA, dnv, bufB, N, D);
        size_t en = (size_t)E * (D >> 2);
        edge_scatter<<<nblk(en, 256), 256, 0, stream>>>(bufA, srcp, dstp, dnv, bufB, E, D);

        zero_f     <<<nblk(512, 256), 256, 0, stream>>>(stat, 512);
        bn_stats   <<<nblk(N, 256), 256, 0, stream>>>(bufB, stat, N, D);
        bn_finalize<<<1, 256, 0, stream>>>(stat, gamma[l], beta[l], bnsc, N, D);
        bn_apply_relu<<<nblk(n4, 256), 256, 0, stream>>>(bufB, bnsc, N, D);
        hin = bufB;
    }

    // --- global mean pool ---
    zero_f  <<<nblk(64 * 256 + 64, 256), 256, 0, stream>>>(psum, 64 * 256 + 64); // psum+pcnt contiguous
    pool_acc<<<nblk(N, 256), 256, 0, stream>>>(bufB, batch, psum, pcnt, N, 256);
    pool_div<<<nblk(64 * 256, 256), 256, 0, stream>>>(psum, pcnt, gbuf, 256);

    // --- FC head (also WMMA; M=64 fits one block row) ---
    { dim3 gg(1, 1024 / TILE_N); gemm_wmma_bf16<<<gg, 256, 0, stream>>>(gbuf, Wf0, bf0, g1, 64, 256, 1024, 1); }
    { dim3 gg(1, 128 / TILE_N);  gemm_wmma_bf16<<<gg, 256, 0, stream>>>(g1, Wf1, bf1, out, 64, 1024, 128, 0); }
}
